// CausalSelfAttention_11536282157232
// MI455X (gfx1250) — matrix operand order repaired and checked by reading, not compiled, not run on hardware
//
#include <hip/hip_runtime.h>
#include <hip/hip_bf16.h>

// ---------------------------------------------------------------------------
// Types for gfx1250 WMMA (wave32): V_WMMA_F32_16X16X32_BF16
// ---------------------------------------------------------------------------
typedef __attribute__((ext_vector_type(16))) __bf16 v16bf;
typedef __attribute__((ext_vector_type(8)))  __bf16 v8bf;
typedef __attribute__((ext_vector_type(8)))  float  v8f;

#define D_MODEL   1024
#define NUM_HEADS 16
#define HEAD_DIM  64
#define SEQ_LEN   2048
#define BATCH     2
#define M_ROWS    (BATCH * SEQ_LEN)   // 4096

static __device__ __forceinline__ unsigned short to_bf16(float f) {
    union { float f; unsigned u; } v; v.f = f;
    unsigned u = v.u;
    return (unsigned short)((u + 0x7FFFu + ((u >> 16) & 1u)) >> 16);
}

static __device__ __forceinline__ v8bf ld8(const unsigned short* p) {
    return *reinterpret_cast<const v8bf*>(p);
}
static __device__ __forceinline__ v16bf ld16(const unsigned short* p) {
    return *reinterpret_cast<const v16bf*>(p);
}
static __device__ __forceinline__ v16bf cat8(v8bf lo, v8bf hi) {
    return __builtin_shufflevector(lo, hi, 0,1,2,3,4,5,6,7,8,9,10,11,12,13,14,15);
}
static __device__ __forceinline__ v8f vzero8() {
    v8f z;
#pragma unroll
    for (int i = 0; i < 8; ++i) z[i] = 0.0f;
    return z;
}
static __device__ __forceinline__ v8f wmma_bf16(v16bf a, v16bf b, v8f c) {
    // (neg_a, A, neg_b, B, c_mod, C, reuse_a, reuse_b)
    return __builtin_amdgcn_wmma_f32_16x16x32_bf16(false, a, false, b, (short)0, c,
                                                   false, false);
}

// Async Global -> LDS copy (gfx1250 ASYNCcnt path).
// LDS dest operand is the wave-relative LDS byte offset; for a __shared__
// generic pointer that is exactly addr[31:0] (LDS aperture lives in the
// high dword), so (unsigned)(size_t)ptr is the offset.
static __device__ __forceinline__ void async_copy_b128(unsigned lds_byte_off,
                                                       const void* gptr) {
    asm volatile("global_load_async_to_lds_b128 %0, %1, off"
                 :: "v"(lds_byte_off),
                    "v"((unsigned long long)(size_t)gptr)
                 : "memory");
}
static __device__ __forceinline__ void wait_async_le2() {
    asm volatile("s_wait_asynccnt 0x2" ::: "memory");
}
static __device__ __forceinline__ void wait_async_0() {
    asm volatile("s_wait_asynccnt 0x0" ::: "memory");
}

// ---------------------------------------------------------------------------
// 1) fp32 -> bf16 cast
// ---------------------------------------------------------------------------
__global__ __launch_bounds__(256) void cvt_f32_to_bf16(const float* __restrict__ in,
                                                       unsigned short* __restrict__ out,
                                                       int n) {
    int i = blockIdx.x * 256 + threadIdx.x;
    if (i < n) out[i] = to_bf16(in[i]);
}

// ---------------------------------------------------------------------------
// 2) GEMM: C[M,N] (f32) = A[M,K] (bf16, row-major) * B[N,K]^T (bf16, row-major)
//    One wave -> 16x64 tile. Ping-pong unroll-by-2 software pipeline: each
//    stage's loads land directly in the registers freed by the previous
//    stage's WMMAs (no register-rotation movs, no VALU WAR-hazard nops —
//    VMEM writes are tracked via LOADcnt). Operands stream from the 192MB L2
//    (the whole working set is L2-resident), so no LDS staging needed.
//    Requires K % 64 == 0 and K >= 64 (true here: K = 1024).
// ---------------------------------------------------------------------------
__global__ __launch_bounds__(256) void gemm_bf16_wmma(const unsigned short* __restrict__ A,
                                                      const unsigned short* __restrict__ B,
                                                      float* __restrict__ C,
                                                      int M, int N, int K) {
    const int lane = threadIdx.x & 31;
    const int warp = threadIdx.x >> 5;
    const int half = lane >> 4;       // 0/1: which 16-lane group
    const int l16  = lane & 15;
    const int m0 = (blockIdx.x * 8 + warp) * 16;
    const int n0 = blockIdx.y * 64;

    v8f acc[4];
#pragma unroll
    for (int t = 0; t < 4; ++t) acc[t] = vzero8();

    const unsigned short* arow = A + (size_t)(m0 + l16) * K;
    const unsigned short* brow[4];
#pragma unroll
    for (int t = 0; t < 4; ++t)
        brow[t] = B + (size_t)(n0 + 16 * t + l16) * K + 16 * half;

    // prologue: stage k = 0 into set A
    v16bf aA = cat8(ld8(arow + 16 * half), ld8(arow + 8 + 16 * half));
    v16bf bA[4];
#pragma unroll
    for (int t = 0; t < 4; ++t) bA[t] = ld16(brow[t]);

    for (int k0 = 0; k0 < K - 64; k0 += 64) {
        // stage k0+32 into set B, consume set A (k0)
        v16bf aB = cat8(ld8(arow + k0 + 32 + 16 * half),
                        ld8(arow + k0 + 40 + 16 * half));
        v16bf bB[4];
#pragma unroll
        for (int t = 0; t < 4; ++t) bB[t] = ld16(brow[t] + k0 + 32);
        __builtin_prefetch(arow + k0 + 256, 0, 3);
#pragma unroll
        for (int t = 0; t < 4; ++t) acc[t] = wmma_bf16(aA, bA[t], acc[t]);

        // stage k0+64 into set A, consume set B (k0+32)
        aA = cat8(ld8(arow + k0 + 64 + 16 * half),
                  ld8(arow + k0 + 72 + 16 * half));
#pragma unroll
        for (int t = 0; t < 4; ++t) bA[t] = ld16(brow[t] + k0 + 64);
#pragma unroll
        for (int t = 0; t < 4; ++t) acc[t] = wmma_bf16(aB, bB[t], acc[t]);
    }
    // epilogue: k = K-64 (in set A) and k = K-32
    {
        v16bf aB = cat8(ld8(arow + K - 32 + 16 * half),
                        ld8(arow + K - 24 + 16 * half));
        v16bf bB[4];
#pragma unroll
        for (int t = 0; t < 4; ++t) bB[t] = ld16(brow[t] + K - 32);
#pragma unroll
        for (int t = 0; t < 4; ++t) acc[t] = wmma_bf16(aA, bA[t], acc[t]);
#pragma unroll
        for (int t = 0; t < 4; ++t) acc[t] = wmma_bf16(aB, bB[t], acc[t]);
    }
    // C layout: reg r -> row r + 8*half, lane%16 -> col
#pragma unroll
    for (int t = 0; t < 4; ++t)
#pragma unroll
        for (int r = 0; r < 8; ++r)
            C[(size_t)(m0 + r + 8 * half) * N + n0 + 16 * t + l16] = acc[t][r];
}

// ---------------------------------------------------------------------------
// 3) RoPE + head split.  qkv f32 [b,s,3072] ->
//    q  bf16 [bh, s, 64]  (rope, pre-scaled by 1/sqrt(64))
//    k  bf16 [bh, s, 64]  (rope)
//    vT bf16 [bh, 64, s]  (transposed so PV's B operand is contiguous)
// ---------------------------------------------------------------------------
__global__ __launch_bounds__(256) void rope_split(const float* __restrict__ qkv,
                                                  unsigned short* __restrict__ qo,
                                                  unsigned short* __restrict__ ko,
                                                  unsigned short* __restrict__ vto) {
    int idx = blockIdx.x * 256 + threadIdx.x;   // [0, 2*16*2048*32)
    int j  = idx & 31;            // rotary pair index
    int s  = (idx >> 5) & 2047;   // sequence position
    int bh = idx >> 16;           // 0..31
    int h = bh & 15, b = bh >> 4;

    const float* base = qkv + ((size_t)(b * SEQ_LEN + s)) * (3 * D_MODEL) + h * HEAD_DIM + 2 * j;
    float q1 = base[0],              q2 = base[1];
    float k1 = base[D_MODEL],        k2 = base[D_MODEL + 1];
    float v1 = base[2 * D_MODEL],    v2 = base[2 * D_MODEL + 1];

    // inv_freq = 10000^(-j/32) = exp(-j * ln(10000)/32)
    float inv = __expf(-0.28782313662425572f * (float)j);
    float ang = (float)s * inv;
    float sn, cs;
    __sincosf(ang, &sn, &cs);

    float qr1 = q1 * cs - q2 * sn, qr2 = q1 * sn + q2 * cs;
    float kr1 = k1 * cs - k2 * sn, kr2 = k1 * sn + k2 * cs;
    const float scale = 0.125f;   // 1/sqrt(HEAD_DIM), folded into q

    size_t qk = ((size_t)bh * SEQ_LEN + s) * HEAD_DIM + 2 * j;
    qo[qk]     = to_bf16(qr1 * scale);
    qo[qk + 1] = to_bf16(qr2 * scale);
    ko[qk]     = to_bf16(kr1);
    ko[qk + 1] = to_bf16(kr2);

    size_t vb = ((size_t)bh * HEAD_DIM + 2 * j) * SEQ_LEN + s;
    vto[vb]           = to_bf16(v1);
    vto[vb + SEQ_LEN] = to_bf16(v2);
}

// ---------------------------------------------------------------------------
// 4) Flash attention, block-cooperative:
//    8 waves/block share one (b,h); K/V tiles for each 32-wide k-step are
//    staged into LDS with async global->LDS copies (double buffered,
//    s_wait_asynccnt + barriers), cutting L2 traffic 8x. Per wave:
//    QK^T = 2 chained K=32 bf16 WMMAs; online softmax via shfl_xor row
//    reductions over the C-layout; P -> bf16 -> LDS -> A-layout; PV: 4 WMMAs.
// ---------------------------------------------------------------------------
__global__ __launch_bounds__(256) void attn_kernel(const unsigned short* __restrict__ Q,
                                                   const unsigned short* __restrict__ Kk,
                                                   const unsigned short* __restrict__ VT,
                                                   unsigned short* __restrict__ Obf) {
    __shared__ alignas(32) unsigned short ldsK[2][32 * HEAD_DIM];  // 2 x 4KB
    __shared__ alignas(32) unsigned short ldsV[2][HEAD_DIM * 32];  // 2 x 4KB
    __shared__ alignas(32) unsigned short ldsP[8][16 * 32];        // 8 x 1KB

    const int tid  = threadIdx.x;
    const int lane = tid & 31;
    const int warp = tid >> 5;
    const int half = lane >> 4;
    const int l16  = lane & 15;
    const int g  = blockIdx.x * 8 + warp;    // 4096 q-tiles total
    const int bh = g >> 7;
    const int q0 = (g & 127) << 4;
    const size_t seq_base = (size_t)bh * SEQ_LEN;
    // block-uniform causal bound: highest q row handled by this block
    const int kmax = ((blockIdx.x & 15) << 7) + 127;

    // --- cooperative staging map: each of 256 threads moves 16B per tile ---
    const int kRow = tid >> 3, kChunk = (tid & 7) * 8;       // K: 32 x 64
    const int vRow = tid >> 2, vChunk = (tid & 3) * 8;       // V^T: 64 x 32
    const unsigned short* kGlob = Kk + (seq_base + kRow) * HEAD_DIM + kChunk;
    const unsigned short* vGlob = VT + ((size_t)bh * HEAD_DIM + vRow) * SEQ_LEN + vChunk;
    unsigned offK[2], offV[2];
#pragma unroll
    for (int bfi = 0; bfi < 2; ++bfi) {
        offK[bfi] = (unsigned)(size_t)&ldsK[bfi][kRow * HEAD_DIM + kChunk];
        offV[bfi] = (unsigned)(size_t)&ldsV[bfi][vRow * 32 + vChunk];
    }

    // Q tile 16x64 in A layout (loaded once, from L2)
    const unsigned short* qrow = Q + (seq_base + q0 + l16) * HEAD_DIM;
    v16bf a0 = cat8(ld8(qrow + 16 * half),      ld8(qrow + 8 + 16 * half));
    v16bf a1 = cat8(ld8(qrow + 32 + 16 * half), ld8(qrow + 40 + 16 * half));

    float mstate[8], lstate[8];
#pragma unroll
    for (int r = 0; r < 8; ++r) { mstate[r] = -3.0e38f; lstate[r] = 0.0f; }
    v8f o[4];
#pragma unroll
    for (int t = 0; t < 4; ++t) o[t] = vzero8();

    unsigned short* myp = &ldsP[warp][0];

    // prologue: stage tile k0 = 0 into buffer 0
    async_copy_b128(offK[0], kGlob);
    async_copy_b128(offV[0], vGlob);

    for (int k0 = 0; k0 <= kmax; k0 += 32) {
        const int cur  = (k0 >> 5) & 1;
        const int k1   = k0 + 32;
        const bool more = (k1 <= kmax);            // block-uniform
        if (more) {
            async_copy_b128(offK[cur ^ 1], kGlob + (size_t)k1 * HEAD_DIM);
            async_copy_b128(offV[cur ^ 1], vGlob + k1);
            wait_async_le2();                      // current tile's 2 copies done
        } else {
            wait_async_0();
        }
        __syncthreads();                           // tile visible to all waves

        if (k0 <= q0 + 15) {                       // wave-uniform causal skip
            const unsigned short* kTile = &ldsK[cur][0];
            const unsigned short* vTile = &ldsV[cur][0];

            // ----- scores: two 16x16 tiles (k cols k0..k0+31) -----
            v8f sc[2];
#pragma unroll
            for (int t = 0; t < 2; ++t) {
                int sctx = k0 + 16 * t + l16;      // this lane's key position
                const unsigned short* krow = kTile + (16 * t + l16) * HEAD_DIM;
                v16bf b0 = ld16(krow + 16 * half);        // d = 0..31
                v16bf b1 = ld16(krow + 32 + 16 * half);   // d = 32..63
                v8f c = vzero8();
                c = wmma_bf16(a0, b0, c);
                c = wmma_bf16(a1, b1, c);
                // causal mask via cndmask (EXEC stays all-1s for WMMA)
#pragma unroll
                for (int r = 0; r < 8; ++r) {
                    int row = q0 + r + 8 * half;
                    c[r] = (sctx > row) ? -3.0e38f : c[r];
                }
                sc[t] = c;
            }

            // ----- online softmax (rows across 16 lanes in one VGPR) -----
            float p0[8], p1[8];
#pragma unroll
            for (int r = 0; r < 8; ++r) {
                float v = fmaxf(sc[0][r], sc[1][r]);
                v = fmaxf(v, __shfl_xor(v, 1, 32));
                v = fmaxf(v, __shfl_xor(v, 2, 32));
                v = fmaxf(v, __shfl_xor(v, 4, 32));
                v = fmaxf(v, __shfl_xor(v, 8, 32));
                float mnew = fmaxf(mstate[r], v);
                float corr = __expf(mstate[r] - mnew);
                mstate[r] = mnew;
                float e0 = __expf(sc[0][r] - mnew);
                float e1 = __expf(sc[1][r] - mnew);
                float rs = e0 + e1;
                rs += __shfl_xor(rs, 1, 32);
                rs += __shfl_xor(rs, 2, 32);
                rs += __shfl_xor(rs, 4, 32);
                rs += __shfl_xor(rs, 8, 32);
                lstate[r] = lstate[r] * corr + rs;
                p0[r] = e0; p1[r] = e1;
#pragma unroll
                for (int t = 0; t < 4; ++t) o[t][r] *= corr;
            }

            // ----- P: C-layout -> LDS -> A-layout (in-order per-wave LDS) ----
#pragma unroll
            for (int r = 0; r < 8; ++r) {
                int row = r + 8 * half;
                myp[row * 32 + l16]      = to_bf16(p0[r]);
                myp[row * 32 + 16 + l16] = to_bf16(p1[r]);
            }
            const unsigned short* prow = myp + l16 * 32;
            v16bf pa = cat8(*reinterpret_cast<const v8bf*>(prow + 16 * half),
                            *reinterpret_cast<const v8bf*>(prow + 8 + 16 * half));

            // ----- PV: out(16x64) += P(16x32) @ V(32x64), V^T tile in LDS ----
#pragma unroll
            for (int t = 0; t < 4; ++t) {
                v16bf vb = ld16(vTile + (16 * t + l16) * 32 + 16 * half);
                o[t] = wmma_bf16(pa, vb, o[t]);
            }
        }
        __syncthreads();     // all waves done reading buffer before overwrite
    }

    // ----- normalize + merge heads: bf16 [b, s, h*64 + d] -----
    const int b = bh >> 4, h = bh & 15;
#pragma unroll
    for (int t = 0; t < 4; ++t)
#pragma unroll
        for (int r = 0; r < 8; ++r) {
            int s = q0 + r + 8 * half;
            float val = o[t][r] / lstate[r];
            Obf[((size_t)(b * SEQ_LEN + s)) * D_MODEL + h * HEAD_DIM + 16 * t + l16] =
                to_bf16(val);
        }
}

// ---------------------------------------------------------------------------
// Host-side orchestration
// ---------------------------------------------------------------------------
extern "C" void kernel_launch(void* const* d_in, const int* in_sizes, int n_in,
                              void* d_out, int out_size, void* d_ws, size_t ws_size,
                              hipStream_t stream) {
    (void)in_sizes; (void)n_in; (void)out_size; (void)ws_size;
    const float* x      = (const float*)d_in[0];   // [2,2048,1024]
    const float* w_qkv  = (const float*)d_in[1];   // [3072,1024]
    const float* w_out  = (const float*)d_in[2];   // [1024,1024]
    float* out = (float*)d_out;                    // [2,2048,1024] fp32

    const int n_x    = M_ROWS * D_MODEL;           // 4,194,304
    const int n_wqkv = 3 * D_MODEL * D_MODEL;      // 3,145,728
    const int n_wout = D_MODEL * D_MODEL;          // 1,048,576
    const int n_head = 2 * NUM_HEADS * SEQ_LEN * HEAD_DIM; // 4,194,304

    char* ws = (char*)d_ws;
    size_t off = 0;
    auto alloc = [&](size_t bytes) {
        void* p = ws + off;
        off += (bytes + 255) & ~(size_t)255;
        return p;
    };
    unsigned short* x_bf    = (unsigned short*)alloc((size_t)n_x * 2);
    unsigned short* wqkv_bf = (unsigned short*)alloc((size_t)n_wqkv * 2);
    unsigned short* wout_bf = (unsigned short*)alloc((size_t)n_wout * 2);
    float*          qkv_f   = (float*)alloc((size_t)M_ROWS * 3 * D_MODEL * 4);
    unsigned short* q_bf    = (unsigned short*)alloc((size_t)n_head * 2);
    unsigned short* k_bf    = (unsigned short*)alloc((size_t)n_head * 2);
    unsigned short* vt_bf   = (unsigned short*)alloc((size_t)n_head * 2);
    unsigned short* attn_bf = (unsigned short*)alloc((size_t)n_x * 2);

    // 1) fp32 -> bf16 casts
    cvt_f32_to_bf16<<<(n_x    + 255) / 256, 256, 0, stream>>>(x,     x_bf,    n_x);
    cvt_f32_to_bf16<<<(n_wqkv + 255) / 256, 256, 0, stream>>>(w_qkv, wqkv_bf, n_wqkv);
    cvt_f32_to_bf16<<<(n_wout + 255) / 256, 256, 0, stream>>>(w_out, wout_bf, n_wout);

    // 2) qkv = x @ w_qkv^T   (M=4096, N=3072, K=1024)
    gemm_bf16_wmma<<<dim3(M_ROWS / 128, 3 * D_MODEL / 64), 256, 0, stream>>>(
        x_bf, wqkv_bf, qkv_f, M_ROWS, 3 * D_MODEL, D_MODEL);

    // 3) RoPE + split heads (+ V transpose)
    rope_split<<<(2 * NUM_HEADS * SEQ_LEN * 32) / 256, 256, 0, stream>>>(
        qkv_f, q_bf, k_bf, vt_bf);

    // 4) causal flash attention (4096 q-tile waves, 8 waves/block, async K/V)
    attn_kernel<<<512, 256, 0, stream>>>(q_bf, k_bf, vt_bf, attn_bf);

    // 5) out = attn @ w_out^T  (M=4096, N=1024, K=1024)
    gemm_bf16_wmma<<<dim3(M_ROWS / 128, D_MODEL / 64), 256, 0, stream>>>(
        attn_bf, wout_bf, out, M_ROWS, D_MODEL, D_MODEL);
}
